// Qwen15MoeSparseMoeBlock_38774964748825
// MI455X (gfx1250) — compile-verified, matrix operand order checked by reading, not run on hardware
//
#include <hip/hip_runtime.h>
#include <hip/hip_bf16.h>

// Problem constants (from reference): B=4,S=1024 -> T=4096 tokens
#define TOK 4096
#define HD  2048
#define NE  16
#define TOPK 4
#define ID  1408
#define SID 5632

typedef __bf16 bf16_t;
typedef __attribute__((ext_vector_type(8)))  bf16_t v8bf;
typedef __attribute__((ext_vector_type(16))) bf16_t v16bf;
typedef __attribute__((ext_vector_type(8)))  float  v8f;
typedef __attribute__((ext_vector_type(4)))  unsigned int v4u;
typedef __attribute__((ext_vector_type(8)))  int v8i_t;
typedef __attribute__((ext_vector_type(4)))  int v4i_t;

// GEMM tile config: block = 256 threads = 8 wave32, tile 128(M) x 64(N) x 64(K)
#define BM 128
#define BN 64
#define BK 64
#define LDT 72   // padded LDS row stride (bf16 elems): 144B = 9*16B, conflict-free frag reads

// NOTE: evaluated per compilation pass; true on the gfx1250 device pass,
// false on the host pass (where the device code is never executed).
#if defined(__has_builtin)
#if __has_builtin(__builtin_amdgcn_tensor_load_to_lds) && __has_builtin(__builtin_amdgcn_s_wait_tensorcnt)
#define HAVE_TDM 1
#endif
#endif
#ifndef HAVE_TDM
#define HAVE_TDM 0
#endif

__device__ __forceinline__ bf16_t f2bf(float f) { return (bf16_t)f; }

__device__ __forceinline__ v8f v8f_zero() {
  v8f z = {0.f, 0.f, 0.f, 0.f, 0.f, 0.f, 0.f, 0.f};
  return z;
}

__device__ __forceinline__ v8f wmma_bf16(v16bf a, v16bf b, v8f c) {
  // (neg_a, A, neg_b, B, c_mod, C, reuse_a, reuse_b) -> v_wmma_f32_16x16x32_bf16
  return __builtin_amdgcn_wmma_f32_16x16x32_bf16(false, a, false, b, (short)0, c, false, false);
}

__device__ __forceinline__ unsigned lds_addr32(const void* p) {
  // Generic pointers into LDS carry the byte offset in the low 32 bits.
  return (unsigned)(size_t)p;
}

// Load a 16x16x32-shaped A/B fragment from an LDS tile stored row-major [rows][LDT].
// Lane layout (ISA 7.12.2, 16-bit A 16x32): lanes 0-15 row=lane, K{0..7,16..23};
// lanes 16-31 row=lane-16, K{8..15,24..31}. B (stored as W rows = B columns) is identical.
__device__ __forceinline__ v16bf frag(const bf16_t* tile, int rowBase, int kk, int lane) {
  int r = lane & 15;
  int h = (lane >> 4) * 8;
  const bf16_t* p = tile + (rowBase + r) * LDT + kk + h;
  v8bf lo = *(const v8bf*)p;
  v8bf hi = *(const v8bf*)(p + 16);
  v16bf o;
#pragma unroll
  for (int i = 0; i < 8; ++i) { o[i] = lo[i]; o[i + 8] = hi[i]; }
  return o;
}

// Issue one TDM descriptor copying a BM x BK bf16 tile (global, row stride `strideElems`)
// into LDS with a 16B pad after every 128B row -> LDT(=72)-element row stride.
// Call from ONE wave only (TDM ignores EXEC; one issue per wave executing it).
// Declared on both passes; body only on the device pass (host never executes it).
__device__ __forceinline__ void tdm_load_a(const bf16_t* gsrc, unsigned ldsAddr,
                                           unsigned dim0Elems, unsigned dim1Rows,
                                           unsigned strideElems) {
#if HAVE_TDM
  unsigned long long ga = (unsigned long long)(size_t)gsrc;
  v4u g0;
  g0[0] = 1u;                                           // count=1, user mode
  g0[1] = ldsAddr;                                      // lds_addr (bytes)
  g0[2] = (unsigned)(ga & 0xFFFFFFFFu);                 // global_addr[31:0]
  g0[3] = (unsigned)((ga >> 32) & 0x1FFFFFFu) | (2u << 30); // global_addr[56:32], type=2
  v8i_t g1;
  // data_size=1 (2B), pad_enable, pad_interval=4 (every 32 DWORDs=128B), pad_amount=3 (4 DWORDs=16B)
  g1[0] = (1 << 16) | (1 << 20) | (4 << 22) | (3 << 25);
  g1[1] = (int)((dim0Elems & 0xFFFFu) << 16);           // atomic_barrier=0 | tensor_dim0[15:0]
  g1[2] = (int)(((dim0Elems >> 16) & 0xFFFFu) | ((dim1Rows & 0xFFFFu) << 16));
  g1[3] = (int)(((dim1Rows >> 16) & 0xFFFFu) | ((unsigned)BK << 16)); // tile_dim0 = 64
  g1[4] = BM;                                           // tile_dim1 = 128, tile_dim2 = 0
  g1[5] = (int)strideElems;                             // tensor_dim0_stride[31:0]
  g1[6] = 0;                                            // stride[47:32] | dim1_stride[15:0]
  g1[7] = 0;
  v4i_t gz4 = {0, 0, 0, 0};
  v8i_t gz8 = {0, 0, 0, 0, 0, 0, 0, 0};
  // amdgpu-toolchain (clang-23) 6-arg form: (g0, g1, g2, g3, g4, cpol)
  __builtin_amdgcn_tensor_load_to_lds(g0, g1, gz4, gz4, gz8, 0);
#else
  (void)gsrc; (void)ldsAddr; (void)dim0Elems; (void)dim1Rows; (void)strideElems;
#endif
}

__device__ __forceinline__ void tdm_wait() {
#if HAVE_TDM
  __builtin_amdgcn_s_wait_tensorcnt(0);
#endif
}

// Fallback / generic A-tile stager (contiguous bf16 rows, clamped rows instead of zero-fill:
// garbage rows are harmless, every C element depends only on its own A row; outputs are guarded).
__device__ __forceinline__ void load_a_vgpr(bf16_t* As, const bf16_t* Ag, long lda,
                                            int mLimit, int tid) {
#pragma unroll
  for (int c = 0; c < 2; ++c) {
    int chunk = tid * 2 + c;          // 512 chunks of 16 bf16
    int row   = chunk >> 2;           // 4 chunks per 64-col row
    int col   = (chunk & 3) * 16;
    int rs    = row < mLimit ? row : (mLimit - 1);
    const v8bf* s = (const v8bf*)(Ag + (long)rs * lda + col);
    v8bf* dst = (v8bf*)&As[row * LDT + col];
    dst[0] = s[0];
    dst[1] = s[1];
  }
}

// Gathered A-tile stager: per-lane async global->LDS copies (ASYNCcnt-tracked).
__device__ __forceinline__ void load_a_async_gather(bf16_t* As, const bf16_t* Xb,
                                                    const int* slots_base, int m0, int cnt,
                                                    int k0, int tid) {
#pragma unroll
  for (int c = 0; c < 2; ++c) {
    int chunk = tid * 2 + c;
    int row   = chunk >> 2;
    int col   = (chunk & 3) * 16;
    int m = m0 + row;
    if (m >= cnt) m = cnt - 1;                 // clamp: garbage rows masked at epilogue
    int tok = slots_base[m];
    const bf16_t* src = Xb + (size_t)tok * HD + k0 + col;
    unsigned lds = lds_addr32(&As[row * LDT + col]);
    unsigned long long ga = (unsigned long long)(size_t)src;
    asm volatile("global_load_async_to_lds_b128 %0, %1, off" :: "v"(lds), "v"(ga) : "memory");
  }
  asm volatile("s_wait_asynccnt 0x0" ::: "memory");
}

// Stage B tile (BN x BK) from fp32 weight rows W[n, k], convert -> bf16 in LDS.
__device__ __forceinline__ void load_b(bf16_t* Bs, const float* Bg, long ldb, int tid) {
  int row = tid >> 2;                 // 0..63
  int col = (tid & 3) * 16;           // 16 floats per thread
  const float4* s = (const float4*)(Bg + (long)row * ldb + col);
  bf16_t* d = &Bs[row * LDT + col];
#pragma unroll
  for (int q = 0; q < 4; ++q) {
    float4 f = s[q];
    d[q * 4 + 0] = f2bf(f.x);
    d[q * 4 + 1] = f2bf(f.y);
    d[q * 4 + 2] = f2bf(f.z);
    d[q * 4 + 3] = f2bf(f.w);
  }
}

// ---------------------------------------------------------------- small kernels

__global__ void k_init(int* counts, int* cursor) {
  int i = threadIdx.x;
  if (i < NE) { counts[i] = 0; cursor[i] = 0; }
}

__global__ void k_convert(const float* __restrict__ x, bf16_t* __restrict__ xb) {
  size_t i = (size_t)blockIdx.x * blockDim.x + threadIdx.x;   // 4 elems each
  float4 f = ((const float4*)x)[i];
  bf16_t* d = xb + i * 4;
  d[0] = f2bf(f.x); d[1] = f2bf(f.y); d[2] = f2bf(f.z); d[3] = f2bf(f.w);
}

__global__ void k_router(const float* __restrict__ x, const float* __restrict__ gw,
                         const float* __restrict__ gvec, float* __restrict__ logits,
                         float* __restrict__ gsig) {
  __shared__ float xs[HD];
  int t = blockIdx.x;
  const float* xr = x + (size_t)t * HD;
  for (int i = threadIdx.x; i < HD; i += 256) xs[i] = xr[i];
  __syncthreads();
  int lane = threadIdx.x & 31, wid = threadIdx.x >> 5;
  for (int e = wid; e < NE + 1; e += 8) {     // e==NE -> shared sigmoid gate vector
    const float* w = (e < NE) ? (gw + (size_t)e * HD) : gvec;
    float acc = 0.f;
    for (int h = lane; h < HD; h += 32) acc += xs[h] * w[h];
#pragma unroll
    for (int o = 16; o > 0; o >>= 1) acc += __shfl_xor(acc, o, 32);
    if (lane == 0) {
      if (e < NE) logits[t * NE + e] = acc;
      else        gsig[t] = 1.f / (1.f + __expf(-acc));
    }
  }
}

__global__ void k_topk(const float* __restrict__ logits, int* __restrict__ tki,
                       float* __restrict__ tkw, int* counts) {
  int t = blockIdx.x * blockDim.x + threadIdx.x;
  if (t >= TOK) return;
  float p[NE];
  float mx = -1e30f;
#pragma unroll
  for (int e = 0; e < NE; ++e) { p[e] = logits[t * NE + e]; mx = fmaxf(mx, p[e]); }
#pragma unroll
  for (int e = 0; e < NE; ++e) p[e] = __expf(p[e] - mx);   // softmax denom cancels in renorm
  float wsum = 0.f;
  int   ids[TOPK]; float ws[TOPK];
#pragma unroll
  for (int j = 0; j < TOPK; ++j) {
    int   bi = 0; float bv = -1.f;
#pragma unroll
    for (int e = 0; e < NE; ++e) if (p[e] > bv) { bv = p[e]; bi = e; }
    ids[j] = bi; ws[j] = bv; wsum += bv; p[bi] = -2.f;
  }
#pragma unroll
  for (int j = 0; j < TOPK; ++j) {
    tki[t * TOPK + j] = ids[j];
    tkw[t * TOPK + j] = ws[j] / wsum;
    atomicAdd(&counts[ids[j]], 1);
  }
}

__global__ void k_scan(const int* __restrict__ counts, int* __restrict__ offsets) {
  if (threadIdx.x == 0) {
    int s = 0;
    for (int e = 0; e < NE; ++e) { offsets[e] = s; s += counts[e]; }
  }
}

__global__ void k_scatter(const int* __restrict__ tki, const float* __restrict__ tkw,
                          const int* __restrict__ offsets, int* cursor,
                          int* __restrict__ slots, float* __restrict__ wgt) {
  int t = blockIdx.x * blockDim.x + threadIdx.x;
  if (t >= TOK) return;
#pragma unroll
  for (int j = 0; j < TOPK; ++j) {
    int e = tki[t * TOPK + j];
    int pos = atomicAdd(&cursor[e], 1);
    int s = offsets[e] + pos;
    slots[s] = t;
    wgt[s] = tkw[t * TOPK + j];
  }
}

// Stage A tile via TDM (wave 0 issues; everyone else overlaps on B conversion).
#define STAGE_A_TDM(As, src, dim0, dim1, stride, mLimit)                         \
  do {                                                                           \
    if (HAVE_TDM) {                                                              \
      if (wid == 0)                                                              \
        tdm_load_a((src), lds_addr32(As), (dim0), (dim1), (stride));             \
    } else {                                                                     \
      load_a_vgpr((As), (src), (stride), (mLimit), tid);                         \
    }                                                                            \
  } while (0)

#define WAIT_A_TDM() do { if (HAVE_TDM && wid == 0) tdm_wait(); } while (0)

// ---------------------------------------------------------------- WMMA GEMMs

// Hs[t, n] = silu(X @ Wg[n].T) * (X @ Wu[n].T), n in [0,SID); fused dual-B tiles.
__global__ void k_shared_gateup(const bf16_t* __restrict__ Xb, const float* __restrict__ Wgu,
                                bf16_t* __restrict__ Hs) {
  __shared__ __align__(16) bf16_t As[BM * LDT];
  __shared__ __align__(16) bf16_t Bg[BN * LDT];
  __shared__ __align__(16) bf16_t Bu[BN * LDT];
  int tid = threadIdx.x, lane = tid & 31, wid = tid >> 5;
  int wm = wid >> 1, wn = wid & 1;
  int m0 = blockIdx.y * BM, n0 = blockIdx.x * BN;
  v8f ag[2][2], au[2][2];
#pragma unroll
  for (int i = 0; i < 2; ++i)
#pragma unroll
    for (int j = 0; j < 2; ++j) { ag[i][j] = v8f_zero(); au[i][j] = v8f_zero(); }

  for (int k0 = 0; k0 < HD; k0 += BK) {
    STAGE_A_TDM(As, Xb + (size_t)m0 * HD + k0, HD, TOK, HD, BM);
    load_b(Bg, Wgu + (size_t)n0 * HD + k0, HD, tid);
    load_b(Bu, Wgu + (size_t)(SID + n0) * HD + k0, HD, tid);
    WAIT_A_TDM();
    __syncthreads();
#pragma unroll
    for (int kk = 0; kk < BK; kk += 32) {
      v16bf a0 = frag(As, wm * 32,      kk, lane);
      v16bf a1 = frag(As, wm * 32 + 16, kk, lane);
      v16bf g0 = frag(Bg, wn * 32,      kk, lane);
      v16bf g1 = frag(Bg, wn * 32 + 16, kk, lane);
      v16bf u0 = frag(Bu, wn * 32,      kk, lane);
      v16bf u1 = frag(Bu, wn * 32 + 16, kk, lane);
      ag[0][0] = wmma_bf16(a0, g0, ag[0][0]);
      ag[0][1] = wmma_bf16(a0, g1, ag[0][1]);
      ag[1][0] = wmma_bf16(a1, g0, ag[1][0]);
      ag[1][1] = wmma_bf16(a1, g1, ag[1][1]);
      au[0][0] = wmma_bf16(a0, u0, au[0][0]);
      au[0][1] = wmma_bf16(a0, u1, au[0][1]);
      au[1][0] = wmma_bf16(a1, u0, au[1][0]);
      au[1][1] = wmma_bf16(a1, u1, au[1][1]);
    }
    __syncthreads();
  }
  int r0 = (lane >> 4) * 8, cn = lane & 15;
#pragma unroll
  for (int mi = 0; mi < 2; ++mi)
#pragma unroll
    for (int ni = 0; ni < 2; ++ni)
#pragma unroll
      for (int r = 0; r < 8; ++r) {
        int row = m0 + wm * 32 + mi * 16 + r0 + r;
        int col = n0 + wn * 32 + ni * 16 + cn;
        float g = ag[mi][ni][r], u = au[mi][ni][r];
        float sv = g / (1.f + __expf(-g));
        Hs[(size_t)row * SID + col] = f2bf(sv * u);
      }
}

// out[t, n] = (Hs @ Wd[n].T) * gsig[t]
__global__ void k_shared_down(const bf16_t* __restrict__ Hs, const float* __restrict__ Wd,
                              const float* __restrict__ gsig, float* __restrict__ out) {
  __shared__ __align__(16) bf16_t As[BM * LDT];
  __shared__ __align__(16) bf16_t Bs[BN * LDT];
  int tid = threadIdx.x, lane = tid & 31, wid = tid >> 5;
  int wm = wid >> 1, wn = wid & 1;
  int m0 = blockIdx.y * BM, n0 = blockIdx.x * BN;
  v8f acc[2][2];
#pragma unroll
  for (int i = 0; i < 2; ++i)
#pragma unroll
    for (int j = 0; j < 2; ++j) acc[i][j] = v8f_zero();

  for (int k0 = 0; k0 < SID; k0 += BK) {
    STAGE_A_TDM(As, Hs + (size_t)m0 * SID + k0, SID, TOK, SID, BM);
    load_b(Bs, Wd + (size_t)n0 * SID + k0, SID, tid);
    WAIT_A_TDM();
    __syncthreads();
#pragma unroll
    for (int kk = 0; kk < BK; kk += 32) {
      v16bf a0 = frag(As, wm * 32,      kk, lane);
      v16bf a1 = frag(As, wm * 32 + 16, kk, lane);
      v16bf b0 = frag(Bs, wn * 32,      kk, lane);
      v16bf b1 = frag(Bs, wn * 32 + 16, kk, lane);
      acc[0][0] = wmma_bf16(a0, b0, acc[0][0]);
      acc[0][1] = wmma_bf16(a0, b1, acc[0][1]);
      acc[1][0] = wmma_bf16(a1, b0, acc[1][0]);
      acc[1][1] = wmma_bf16(a1, b1, acc[1][1]);
    }
    __syncthreads();
  }
  int r0 = (lane >> 4) * 8, cn = lane & 15;
#pragma unroll
  for (int mi = 0; mi < 2; ++mi)
#pragma unroll
    for (int ni = 0; ni < 2; ++ni)
#pragma unroll
      for (int r = 0; r < 8; ++r) {
        int row = m0 + wm * 32 + mi * 16 + r0 + r;
        int col = n0 + wn * 32 + ni * 16 + cn;
        out[(size_t)row * HD + col] = acc[mi][ni][r] * gsig[row];
      }
}

// Per-expert gathered fused gate_up: He[off+m, n] = silu(.)*(.)
__global__ void k_expert_gateup(const bf16_t* __restrict__ Xb, const float* __restrict__ Wgu_all,
                                const int* __restrict__ counts, const int* __restrict__ offsets,
                                const int* __restrict__ slots, bf16_t* __restrict__ He) {
  int e = blockIdx.z;
  int cnt = counts[e];
  int m0 = blockIdx.y * BM;
  if (m0 >= cnt) return;
  int off = offsets[e];
  const float* Wgu = Wgu_all + (size_t)e * 2 * ID * HD;
  __shared__ __align__(16) bf16_t As[BM * LDT];
  __shared__ __align__(16) bf16_t Bg[BN * LDT];
  __shared__ __align__(16) bf16_t Bu[BN * LDT];
  int tid = threadIdx.x, lane = tid & 31, wid = tid >> 5;
  int wm = wid >> 1, wn = wid & 1;
  int n0 = blockIdx.x * BN;
  v8f ag[2][2], au[2][2];
#pragma unroll
  for (int i = 0; i < 2; ++i)
#pragma unroll
    for (int j = 0; j < 2; ++j) { ag[i][j] = v8f_zero(); au[i][j] = v8f_zero(); }

  for (int k0 = 0; k0 < HD; k0 += BK) {
    load_a_async_gather(As, Xb, slots + off, m0, cnt, k0, tid);
    load_b(Bg, Wgu + (size_t)n0 * HD + k0, HD, tid);
    load_b(Bu, Wgu + (size_t)(ID + n0) * HD + k0, HD, tid);
    __syncthreads();
#pragma unroll
    for (int kk = 0; kk < BK; kk += 32) {
      v16bf a0 = frag(As, wm * 32,      kk, lane);
      v16bf a1 = frag(As, wm * 32 + 16, kk, lane);
      v16bf g0 = frag(Bg, wn * 32,      kk, lane);
      v16bf g1 = frag(Bg, wn * 32 + 16, kk, lane);
      v16bf u0 = frag(Bu, wn * 32,      kk, lane);
      v16bf u1 = frag(Bu, wn * 32 + 16, kk, lane);
      ag[0][0] = wmma_bf16(a0, g0, ag[0][0]);
      ag[0][1] = wmma_bf16(a0, g1, ag[0][1]);
      ag[1][0] = wmma_bf16(a1, g0, ag[1][0]);
      ag[1][1] = wmma_bf16(a1, g1, ag[1][1]);
      au[0][0] = wmma_bf16(a0, u0, au[0][0]);
      au[0][1] = wmma_bf16(a0, u1, au[0][1]);
      au[1][0] = wmma_bf16(a1, u0, au[1][0]);
      au[1][1] = wmma_bf16(a1, u1, au[1][1]);
    }
    __syncthreads();
  }
  int r0 = (lane >> 4) * 8, cn = lane & 15;
#pragma unroll
  for (int mi = 0; mi < 2; ++mi)
#pragma unroll
    for (int ni = 0; ni < 2; ++ni)
#pragma unroll
      for (int r = 0; r < 8; ++r) {
        int m = m0 + wm * 32 + mi * 16 + r0 + r;
        if (m < cnt) {
          int col = n0 + wn * 32 + ni * 16 + cn;
          float g = ag[mi][ni][r], u = au[mi][ni][r];
          float sv = g / (1.f + __expf(-g));
          He[(size_t)(off + m) * ID + col] = f2bf(sv * u);
        }
      }
}

// Per-expert down proj + weighted scatter-add into d_out.
__global__ void k_expert_down(const bf16_t* __restrict__ He, const float* __restrict__ Wd_all,
                              const int* __restrict__ counts, const int* __restrict__ offsets,
                              const int* __restrict__ slots, const float* __restrict__ wgt,
                              float* __restrict__ out) {
  int e = blockIdx.z;
  int cnt = counts[e];
  int m0 = blockIdx.y * BM;
  if (m0 >= cnt) return;
  int off = offsets[e];
  const float* Wd = Wd_all + (size_t)e * HD * ID;
  __shared__ __align__(16) bf16_t As[BM * LDT];
  __shared__ __align__(16) bf16_t Bs[BN * LDT];
  int tid = threadIdx.x, lane = tid & 31, wid = tid >> 5;
  int wm = wid >> 1, wn = wid & 1;
  int n0 = blockIdx.x * BN;
  v8f acc[2][2];
#pragma unroll
  for (int i = 0; i < 2; ++i)
#pragma unroll
    for (int j = 0; j < 2; ++j) acc[i][j] = v8f_zero();

  for (int k0 = 0; k0 < ID; k0 += BK) {
    // He rows are contiguous per expert; He has BM rows of slack so clamping is unnecessary.
    STAGE_A_TDM(As, He + (size_t)(off + m0) * ID + k0, ID, TOK * TOPK + BM, ID, cnt - m0);
    load_b(Bs, Wd + (size_t)n0 * ID + k0, ID, tid);
    WAIT_A_TDM();
    __syncthreads();
#pragma unroll
    for (int kk = 0; kk < BK; kk += 32) {
      v16bf a0 = frag(As, wm * 32,      kk, lane);
      v16bf a1 = frag(As, wm * 32 + 16, kk, lane);
      v16bf b0 = frag(Bs, wn * 32,      kk, lane);
      v16bf b1 = frag(Bs, wn * 32 + 16, kk, lane);
      acc[0][0] = wmma_bf16(a0, b0, acc[0][0]);
      acc[0][1] = wmma_bf16(a0, b1, acc[0][1]);
      acc[1][0] = wmma_bf16(a1, b0, acc[1][0]);
      acc[1][1] = wmma_bf16(a1, b1, acc[1][1]);
    }
    __syncthreads();
  }
  int r0 = (lane >> 4) * 8, cn = lane & 15;
#pragma unroll
  for (int mi = 0; mi < 2; ++mi)
#pragma unroll
    for (int ni = 0; ni < 2; ++ni)
#pragma unroll
      for (int r = 0; r < 8; ++r) {
        int m = m0 + wm * 32 + mi * 16 + r0 + r;
        if (m < cnt) {
          int s = off + m;
          int tokid = slots[s];
          float w = wgt[s];
          int col = n0 + wn * 32 + ni * 16 + cn;
          atomicAdd(&out[(size_t)tokid * HD + col], acc[mi][ni][r] * w);
        }
      }
}

// ---------------------------------------------------------------- launcher

extern "C" void kernel_launch(void* const* d_in, const int* in_sizes, int n_in,
                              void* d_out, int out_size, void* d_ws, size_t ws_size,
                              hipStream_t stream) {
  const float* x   = (const float*)d_in[0];   // [T, H]
  const float* gw  = (const float*)d_in[1];   // [E, H]
  const float* egu = (const float*)d_in[2];   // [E, 2I, H]
  const float* ed  = (const float*)d_in[3];   // [E, H, I]
  const float* sgu = (const float*)d_in[4];   // [2SI, H]
  const float* sd  = (const float*)d_in[5];   // [H, SI]
  const float* gv  = (const float*)d_in[6];   // [1, H]
  float* out = (float*)d_out;                 // [T, H]

  char* ws = (char*)d_ws;
  size_t o = 0;
  auto take = [&](size_t bytes) { char* p = ws + o; o += (bytes + 255) & ~(size_t)255; return p; };
  bf16_t* Xb     = (bf16_t*)take((size_t)TOK * HD * 2);       // 16 MB
  float*  logits = (float*) take((size_t)TOK * NE * 4);
  int*    tki    = (int*)   take((size_t)TOK * TOPK * 4);
  float*  tkw    = (float*) take((size_t)TOK * TOPK * 4);
  float*  gsig   = (float*) take((size_t)TOK * 4);
  int*    counts = (int*)   take(NE * 4);
  int*    cursor = (int*)   take(NE * 4);
  int*    offs   = (int*)   take(NE * 4);
  int*    slots  = (int*)   take((size_t)TOK * TOPK * 4);
  float*  wgt    = (float*) take((size_t)TOK * TOPK * 4);
  bf16_t* Hs     = (bf16_t*)take((size_t)TOK * SID * 2);              // 44 MB
  bf16_t* He     = (bf16_t*)take((size_t)(TOK * TOPK + BM) * ID * 2); // 44 MB (+slack rows)
  (void)ws_size; (void)in_sizes; (void)n_in; (void)out_size;  // total ~105 MB

  k_init<<<1, 32, 0, stream>>>(counts, cursor);
  k_convert<<<(TOK * HD / 4) / 256, 256, 0, stream>>>(x, Xb);
  k_router<<<TOK, 256, 0, stream>>>(x, gw, gv, logits, gsig);
  k_topk<<<TOK / 256, 256, 0, stream>>>(logits, tki, tkw, counts);
  k_scan<<<1, 32, 0, stream>>>(counts, offs);
  k_scatter<<<TOK / 256, 256, 0, stream>>>(tki, tkw, offs, cursor, slots, wgt);

  // Shared expert
  k_shared_gateup<<<dim3(SID / BN, TOK / BM), 256, 0, stream>>>(Xb, sgu, Hs);
  k_shared_down  <<<dim3(HD / BN, TOK / BM), 256, 0, stream>>>(Hs, sd, gsig, out);

  // Sparse experts (worst-case static grid; tiles beyond per-expert count exit early)
  k_expert_gateup<<<dim3(ID / BN, TOK / BM, NE), 256, 0, stream>>>(Xb, egu, counts, offs, slots, He);
  k_expert_down  <<<dim3(HD / BN, TOK / BM, NE), 256, 0, stream>>>(He, ed, counts, offs, slots, wgt, out);
}
